// AnchorMlgDml_75737453298235
// MI455X (gfx1250) — matrix operand order repaired and checked by reading, not compiled, not run on hardware
//
#include <hip/hip_runtime.h>
#include <math.h>

// ---------- CDNA5 WMMA types ----------
typedef __attribute__((ext_vector_type(8)))  __bf16 v8bf;
typedef __attribute__((ext_vector_type(16))) __bf16 v16bf;
typedef __attribute__((ext_vector_type(8)))  float  v8f;

#define WMMA_BF16(a, b, c) \
  __builtin_amdgcn_wmma_f32_16x16x32_bf16(false, (a), false, (b), (short)0, (c), false, false)

// ---------- constants ----------
#define NROWS       65536
#define D_IN        128
#define MLG_SIZE    256
#define SAMPLE_SIZE 1024
#define D_OUT       10

// workspace layout (bytes)
#define WS_ANCHB 0          // 1024*128 bf16 = 262144
#define WS_A2    262144     // 1024 f32     = 4096
#define WS_W1B   266240     // 128*128 bf16 = 32768
#define WS_W2B   299008     // 128*128 bf16 = 32768
#define WS_WCF   331776     // 1024*16 bf16 = 32768 (pre-swizzled Wc^T B-fragments)

// ---------- fast activations (1 trans + 1 rcp each, branchless) ----------
__device__ __forceinline__ float fast_rcp(float x) { return __builtin_amdgcn_rcpf(x); }

// mish(x) = x * tanh(softplus(x)); with t=e^x: tanh(sp(x)) = (t^2+2t)/(t^2+2t+2)
__device__ __forceinline__ float mishf(float x) {
  float t = __expf(fminf(x, 20.0f));
  float n = t * (t + 2.0f);
  return x * n * fast_rcp(n + 2.0f);
}
// tanh(y) = 1 - 2/(e^{2y}+1)
__device__ __forceinline__ float tanh_fast(float y) {
  float t = __expf(fminf(2.0f * y, 40.0f));
  return 1.0f - 2.0f * fast_rcp(t + 1.0f);
}

// ---------- async global->LDS copy (gfx1250 GLOBAL_LOAD_ASYNC_TO_LDS_B128) ----------
#if defined(__has_builtin)
#if __has_builtin(__builtin_amdgcn_global_load_async_to_lds_b128)
#define HAVE_ASYNC_LDS 1
#endif
#endif
#ifndef HAVE_ASYNC_LDS
#define HAVE_ASYNC_LDS 0
#endif

#if HAVE_ASYNC_LDS
// exact parameter types per hipcc diagnostic: int __attribute__((vector_size(16))) __device__ *
typedef int v4i_ __attribute__((vector_size(16)));
typedef __attribute__((address_space(1))) v4i_ g_v4i;
typedef __attribute__((address_space(3))) v4i_ l_v4i;
#endif

// copy 32 KB (128x128 bf16) global -> LDS with 256 threads
__device__ __forceinline__ void copy_w_to_lds(__bf16* dst, const __bf16* src, int tid) {
#if HAVE_ASYNC_LDS
#pragma unroll
  for (int i = 0; i < 8; ++i) {
    int c = tid + i * 256;
    __builtin_amdgcn_global_load_async_to_lds_b128(
        (g_v4i*)(src + c * 8), (l_v4i*)(dst + c * 8), 0, 0);
  }
#if __has_builtin(__builtin_amdgcn_s_wait_asynccnt)
  __builtin_amdgcn_s_wait_asynccnt(0);
#else
  asm volatile("s_wait_asynccnt 0x0" ::: "memory");
#endif
#else
#pragma unroll
  for (int i = 0; i < 8; ++i) {
    int c = tid + i * 256;
    ((uint4*)dst)[c] = ((const uint4*)src)[c];
  }
#endif
}

// ---------- fragment loaders (per CDNA5 ISA 16-bit layouts) ----------
__device__ __forceinline__ v16bf frag_join(v8bf lo, v8bf hi) {
  v16bf r;
#pragma unroll
  for (int i = 0; i < 8; ++i) { r[i] = lo[i]; r[i + 8] = hi[i]; }
  return r;
}
__device__ __forceinline__ v16bf frag_pair(const __bf16* p0, const __bf16* p1) {
  return frag_join(*(const v8bf*)p0, *(const v8bf*)p1);
}
__device__ __forceinline__ v16bf frag_contig(const __bf16* p) {
  return frag_pair(p, p + 8);
}
// A fragment 16x32: row m = m0 + lane%16; K pairs at (8*half + 0..7) and (16 + 8*half + 0..7)
__device__ __forceinline__ v16bf loadA(const __bf16* base, int ld, int m0, int kblk, int lane) {
  int m = m0 + (lane & 15), h = lane >> 4;
  const __bf16* p = base + m * ld + kblk * 32;
  return frag_pair(p + 8 * h, p + 16 + 8 * h);
}
// B fragment 32x16 from row-major src[n][K]: col n = n0 + lane%16; K pairs at (8*half + 0..7) and (16 + 8*half + 0..7)
__device__ __forceinline__ v16bf loadB(const __bf16* rm, int ld, int n0, int kblk, int lane) {
  int n = n0 + (lane & 15), h = lane >> 4;
  const __bf16* p = rm + n * ld + kblk * 32;
  return frag_pair(p + 8 * h, p + 16 + 8 * h);
}

// ============================================================================
// Kernel 1: anchors = encode_dml(tanh(Wa @ mlg)) -> bf16 [1024][128], a2[1024]
// ============================================================================
__global__ void anchor_prep(const float* __restrict__ Wa, const float* __restrict__ mlg,
                            const float* __restrict__ W1, const float* __restrict__ b1,
                            const float* __restrict__ W2, const float* __restrict__ b2,
                            __bf16* __restrict__ anchb, float* __restrict__ a2f) {
  __shared__ float t0[128];
  __shared__ float t1[128];
  int s = blockIdx.x, d = threadIdx.x;
  float acc = 0.f;
  for (int k = 0; k < MLG_SIZE; ++k)
    acc = fmaf(Wa[s * MLG_SIZE + k], mlg[k * D_IN + d], acc);
  t0[d] = tanh_fast(acc);
  __syncthreads();
  float h = b1[d];
  for (int j = 0; j < D_IN; ++j) h = fmaf(t0[j], W1[d * D_IN + j], h);
  t1[d] = mishf(h);
  __syncthreads();
  float o = b2[d];
  for (int j = 0; j < D_IN; ++j) o = fmaf(t1[j], W2[d * D_IN + j], o);
  o = mishf(o);
  __bf16 ob = (__bf16)o;
  float of = (float)ob;
  anchb[s * D_IN + d] = ob;
  t0[d] = of * of;
  __syncthreads();
  if (d == 0) {
    float r = 0.f;
    for (int j = 0; j < D_IN; ++j) r += t0[j];
    a2f[s] = r;
  }
}

// ============================================================================
// Kernel 2: weights -> bf16; Wc^T zero-padded & pre-swizzled into B-fragment order
// ============================================================================
__global__ void weight_prep(const float* __restrict__ W1, const float* __restrict__ W2,
                            const float* __restrict__ Wc,
                            __bf16* __restrict__ w1b, __bf16* __restrict__ w2b,
                            __bf16* __restrict__ wcf) {
  int i = blockIdx.x * 256 + threadIdx.x;
  if (i < 16384) {
    w1b[i] = (__bf16)W1[i];
  } else if (i < 32768) {
    int j = i - 16384;
    w2b[j] = (__bf16)W2[j];
  } else if (i < 49152) {
    int j = i - 32768;                 // wcf[((sb*2+h)*16 + c)*16 + e]
    int e = j & 15, c = (j >> 4) & 15, h = (j >> 8) & 1, sb = j >> 9;
    int k = sb * 32 + (e < 8 ? 8 * h + e : 16 + 8 * h + (e - 8));      // anchor index (K of logits GEMM)
    wcf[j] = (c < D_OUT) ? (__bf16)Wc[c * SAMPLE_SIZE + k] : (__bf16)0.0f;
  }
}

// ============================================================================
// Kernel 3: fused encode + pairwise-L2 + logits + log_softmax (all-WMMA)
//   block = 256 threads (8 waves), 128 x-rows per block
//   dynamic LDS: sA(32K) sH(32K) sW(32K) stage(8K) sX2(512) = 104.5 KB
// ============================================================================
#define LDS_BYTES (32768 * 3 + 8192 + 512)

__global__ void __launch_bounds__(256)
fused_main(const float* __restrict__ x,
           const float* __restrict__ b1g, const float* __restrict__ b2g,
           const float* __restrict__ bcg,
           const __bf16* __restrict__ w1b, const __bf16* __restrict__ w2b,
           const __bf16* __restrict__ anchb, const float* __restrict__ a2f,
           const __bf16* __restrict__ wcf, float* __restrict__ out) {
  extern __shared__ char smem[];
  __bf16* sA    = (__bf16*)smem;                  // x tile, later x_dml (bf16, 128x128)
  __bf16* sH    = (__bf16*)(smem + 32768);        // hidden (bf16, 128x128)
  __bf16* sW    = (__bf16*)(smem + 65536);        // W1 then W2 (bf16, 128x128)
  __bf16* stage = (__bf16*)(smem + 98304);        // per-wave 16x32 bf16 re-layout buffers
  float*  sX2   = (float*)(smem + 106496);        // per-row ||x_dml||^2

  const int tid  = threadIdx.x;
  const int wv   = tid >> 5;
  const int lane = tid & 31;
  const int colL = lane & 15;
  const int half = lane >> 4;
  const int m0   = wv * 16;                       // this wave's 16-row stripe
  const int rb   = blockIdx.x * 128;              // block's first x row

  // ---- cooperative: W1 (async DMA to LDS) + x tile (f32 -> bf16) + zero x2 ----
  copy_w_to_lds(sW, w1b, tid);
  {
    const float* xg = x + (size_t)rb * D_IN;
    for (int i = tid; i < 128 * 32; i += 256) {   // float4 chunks
      int row = i >> 5, c4 = (i & 31) * 4;
      float4 v = *(const float4*)(xg + row * D_IN + c4);
      __bf16 e0 = (__bf16)v.x, e1 = (__bf16)v.y, e2 = (__bf16)v.z, e3 = (__bf16)v.w;
      unsigned u0 = (unsigned)__builtin_bit_cast(unsigned short, e0) |
                    ((unsigned)__builtin_bit_cast(unsigned short, e1) << 16);
      unsigned u1 = (unsigned)__builtin_bit_cast(unsigned short, e2) |
                    ((unsigned)__builtin_bit_cast(unsigned short, e3) << 16);
      uint2 pk; pk.x = u0; pk.y = u1;
      *(uint2*)(sA + row * D_IN + c4) = pk;
    }
    if (tid < 128) sX2[tid] = 0.f;
  }
  __syncthreads();

  // ---- stage E1: sH = mish(x @ W1^T + b1) ----
  for (int jt = 0; jt < 8; ++jt) {
    v8f c = {};
#pragma unroll
    for (int kb = 0; kb < 4; ++kb) {
      v16bf a = loadA(sA, D_IN, m0, kb, lane);
      v16bf b = loadB(sW, D_IN, jt * 16, kb, lane);
      c = WMMA_BF16(a, b, c);
    }
    int col = jt * 16 + colL;
    float bias = b1g[col];
#pragma unroll
    for (int r = 0; r < 8; ++r) {
      int row = m0 + r + 8 * half;
      sH[row * D_IN + col] = (__bf16)mishf(c[r] + bias);
    }
  }
  __syncthreads();

  // ---- swap in W2 (async DMA to LDS) ----
  copy_w_to_lds(sW, w2b, tid);
  __syncthreads();

  // ---- stage E2: sA = mish(h @ W2^T + b2), accumulate per-row ||.||^2 ----
  {
    float x2acc[8];
#pragma unroll
    for (int r = 0; r < 8; ++r) x2acc[r] = 0.f;
    for (int jt = 0; jt < 8; ++jt) {
      v8f c = {};
#pragma unroll
      for (int kb = 0; kb < 4; ++kb) {
        v16bf a = loadA(sH, D_IN, m0, kb, lane);
        v16bf b = loadB(sW, D_IN, jt * 16, kb, lane);
        c = WMMA_BF16(a, b, c);
      }
      int col = jt * 16 + colL;
      float bias = b2g[col];
#pragma unroll
      for (int r = 0; r < 8; ++r) {
        int row = m0 + r + 8 * half;
        float v = mishf(c[r] + bias);
        __bf16 vb = (__bf16)v;
        float vf = (float)vb;
        sA[row * D_IN + col] = vb;
        x2acc[r] += vf * vf;
      }
    }
#pragma unroll
    for (int r = 0; r < 8; ++r)
      atomicAdd(&sX2[m0 + r + 8 * half], x2acc[r]);
  }
  __syncthreads();

  // ---- distance + logits stage (per-wave 16-row stripe over 1024 anchors) ----
  float x2v[8];
#pragma unroll
  for (int r = 0; r < 8; ++r) x2v[r] = sX2[m0 + r + 8 * half];

  v16bf axf[4];                                   // hoist x_dml A fragments (K=128)
#pragma unroll
  for (int kb = 0; kb < 4; ++kb) axf[kb] = loadA(sA, D_IN, m0, kb, lane);

  __bf16* stw = stage + wv * 512;                 // 16x32 bf16 per wave
  v8f lc = {};                                    // logits accumulator (16 rows x 16 cols)

  for (int sb = 0; sb < 32; ++sb) {
#pragma unroll
    for (int t = 0; t < 2; ++t) {
      int s0 = sb * 32 + t * 16;
      v8f dc = {};
#pragma unroll
      for (int kb = 0; kb < 4; ++kb) {
        v16bf b = loadB(anchb, D_IN, s0, kb, lane);
        dc = WMMA_BF16(axf[kb], b, dc);
      }
      __builtin_prefetch(anchb + (size_t)(s0 + 32 + colL) * D_IN, 0, 0);
      float a2v = a2f[s0 + colL];
#pragma unroll
      for (int r = 0; r < 8; ++r) {
        float sq = x2v[r] + a2v - 2.f * dc[r];
        float d  = sqrtf(fmaxf(sq, 0.f));
        stw[(r + 8 * half) * 32 + t * 16 + colL] = (__bf16)d;   // C-frag -> A-frag re-layout
      }
    }
    // contract 16x32 distance tile against pre-swizzled Wc^T fragment
    v16bf ad = loadA(stw, 32, 0, 0, lane);        // same-wave LDS ops are in-order
    v16bf bw = frag_contig(wcf + (((sb * 2) + half) * 16 + colL) * 16);
    lc = WMMA_BF16(ad, bw, lc);
  }

  // ---- epilogue: +bc, log_softmax over 10 classes, write out ----
  {
    float* stf = (float*)stw;                     // reuse 1 KB per-wave staging as f32 16x16
#pragma unroll
    for (int r = 0; r < 8; ++r) {
      float v = lc[r] + (colL < D_OUT ? bcg[colL] : 0.f);
      stf[(r + 8 * half) * 16 + colL] = v;
    }
    __syncthreads();
    if (lane < 16) {
      int row = rb + m0 + lane;
      float f[D_OUT];
      float mx = -3.4e38f;
#pragma unroll
      for (int c = 0; c < D_OUT; ++c) { f[c] = stf[lane * 16 + c]; mx = fmaxf(mx, f[c]); }
      float se = 0.f;
#pragma unroll
      for (int c = 0; c < D_OUT; ++c) se += __expf(f[c] - mx);
      float lse = mx + __logf(se);
#pragma unroll
      for (int c = 0; c < D_OUT; ++c) out[row * D_OUT + c] = f[c] - lse;
    }
  }
}

// ============================================================================
extern "C" void kernel_launch(void* const* d_in, const int* in_sizes, int n_in,
                              void* d_out, int out_size, void* d_ws, size_t ws_size,
                              hipStream_t stream) {
  (void)in_sizes; (void)n_in; (void)out_size; (void)ws_size;
  const float* x   = (const float*)d_in[0];
  const float* mlg = (const float*)d_in[1];
  const float* W1  = (const float*)d_in[2];
  const float* b1  = (const float*)d_in[3];
  const float* W2  = (const float*)d_in[4];
  const float* b2  = (const float*)d_in[5];
  const float* Wa  = (const float*)d_in[6];
  const float* Wc  = (const float*)d_in[7];
  const float* bc  = (const float*)d_in[8];
  float* out = (float*)d_out;

  char* ws = (char*)d_ws;
  __bf16* anchb = (__bf16*)(ws + WS_ANCHB);
  float*  a2f   = (float*)(ws + WS_A2);
  __bf16* w1b   = (__bf16*)(ws + WS_W1B);
  __bf16* w2b   = (__bf16*)(ws + WS_W2B);
  __bf16* wcf   = (__bf16*)(ws + WS_WCF);

  anchor_prep<<<SAMPLE_SIZE, 128, 0, stream>>>(Wa, mlg, W1, b1, W2, b2, anchb, a2f);
  weight_prep<<<192, 256, 0, stream>>>(W1, W2, Wc, w1b, w2b, wcf);
  fused_main<<<NROWS / 128, 256, LDS_BYTES, stream>>>(
      x, b1, b2, bc, w1b, w2b, anchb, a2f, wcf, out);
}
